// CIA_56049323213169
// MI455X (gfx1250) — matrix operand order repaired and checked by reading, not compiled, not run on hardware
//
#include <hip/hip_runtime.h>

// ---------------------------------------------------------------------------
// Problem constants (reference: B=4, C=1024, H=64, W=64)
//   N  = B*H*W = 16384, total elements = 16,777,216 (64 MiB f32)
// ---------------------------------------------------------------------------
#define DIM_C   1024
#define DIM_N   16384
#define TOTAL   16777216   // C * N
#define ROWS_W  262144     // B*C*H rows of length W=64

typedef __attribute__((ext_vector_type(16))) __bf16 v16bf;
typedef __attribute__((ext_vector_type(8)))  float  v8f;
typedef __attribute__((ext_vector_type(4)))  unsigned int v4u;
typedef __attribute__((ext_vector_type(8)))  int  v8i;
typedef __attribute__((ext_vector_type(4)))  int  v4i;

union FragBF {
    v16bf v;
    uint4 q[2];
    unsigned short s[16];
};

__device__ __forceinline__ unsigned short bf16bits(float f) {
    unsigned u = __builtin_bit_cast(unsigned, f);
    u += 0x7FFFu + ((u >> 16) & 1u);           // round-to-nearest-even
    return (unsigned short)(u >> 16);
}

__device__ __forceinline__ float waveRedMax(float v) {
#pragma unroll
    for (int o = 16; o > 0; o >>= 1) v = fmaxf(v, __shfl_xor(v, o, 32));
    return v;
}
__device__ __forceinline__ float waveRedSum(float v) {
#pragma unroll
    for (int o = 16; o > 0; o >>= 1) v += __shfl_xor(v, o, 32);
    return v;
}

// ---------------------------------------------------------------------------
// TDM: async 2-D tile load global->LDS (Tensor Data Mover, TENSORcnt-tracked).
// Descriptor packed per CDNA5 D# layout (ISA ch.10.7/8):
//   g0: count=1 | lds_addr[63:32] | global_addr[120:64] | type=2[127:126]
//   g1: data_size=1(2B)[17:16] | pad_enable[20] | pad_interval[24:22] |
//       pad_amount[31:25] | tensor_dim0[79:48] | tensor_dim1[111:80] |
//       tile_dim0[127:112] | tile_dim1[143:128] | dim0_stride[207:160]
//   groups 2/3 zero -> 2-D tensor. This toolchain's builtin is the 6-arg
//   (clang-23 / therock-10.0) form with a trailing int32x8 group.
// ---------------------------------------------------------------------------
__device__ __forceinline__ void tdm_load_tile_2d(
    unsigned lds_byte_addr, const unsigned short* gsrc,
    unsigned tensor_d0, unsigned long long stride0_elems,
    unsigned tile_d0, unsigned tile_d1,
    unsigned pad_interval_code, unsigned pad_amount_code)
{
    unsigned long long ga = (unsigned long long)(size_t)gsrc;
    v4u g0;
    g0.x = 1u;                                           // count=1
    g0.y = lds_byte_addr;                                // LDS dest (bytes)
    g0.z = (unsigned)ga;                                 // global addr lo
    g0.w = (unsigned)((ga >> 32) & 0x01FFFFFFull)        // global addr hi
         | (2u << 30);                                   // type=2 ("image")
    const unsigned td1 = 1u << 20;                       // generous OOB bound
    v8i g1;
    g1[0] = (int)((1u << 16)                             // data_size = 2B
                | (1u << 20)                             // pad_enable
                | (pad_interval_code << 22)
                | (pad_amount_code << 25));
    g1[1] = (int)((tensor_d0 & 0xFFFFu) << 16);
    g1[2] = (int)((tensor_d0 >> 16) | ((td1 & 0xFFFFu) << 16));
    g1[3] = (int)((td1 >> 16) | (tile_d0 << 16));
    g1[4] = (int)(tile_d1 & 0xFFFFu);                    // tile_dim2 = 0
    g1[5] = (int)(unsigned)(stride0_elems & 0xFFFFFFFFull);
    g1[6] = (int)(unsigned)((stride0_elems >> 32) & 0xFFFFull);
    g1[7] = 0;
    v4i z4 = {0, 0, 0, 0};
    v8i z8 = {0, 0, 0, 0, 0, 0, 0, 0};
    __builtin_amdgcn_tensor_load_to_lds(g0, g1, z4, z4, z8, 0);
}

// ---------------------------------------------------------------------------
// Kernel 0: f32 -> bf16 convert (one pass; GEMM operand set becomes 32 MiB,
// fully resident in the 192 MiB L2)
// ---------------------------------------------------------------------------
__global__ void CIA_cvt_bf16(const float* __restrict__ x,
                             unsigned short* __restrict__ y) {
    size_t i = ((size_t)blockIdx.x * blockDim.x + threadIdx.x) * 4;
    float4 v = *(const float4*)(x + i);
    uint2 o;
    o.x = (unsigned)bf16bits(v.x) | ((unsigned)bf16bits(v.y) << 16);
    o.y = (unsigned)bf16bits(v.z) | ((unsigned)bf16bits(v.w) << 16);
    *(uint2*)(y + i) = o;
}

// ---------------------------------------------------------------------------
// bf16 WMMA GEMM with TDM double-buffered staging:
//   C(f32, MxN) = A(bf16, MxK, lda) * B(bf16, KxN, ldb)
// 256 thr = 8 waves; block tile 128x128; wave tile 32x64 = 2x4 WMMA accs.
// Wave 0 issues two tensor_load_to_lds per K-step (A tile 128x32, B tile
// 32x128, both row-major with HW pad for LDS bank spread), waits TENSORcnt,
// then a workgroup barrier publishes the tiles.
// ---------------------------------------------------------------------------
#define BM 128
#define BN 128
#define BK 32
#define SA_LD (BK + 8)   // 40 halves/row (64B data + 16B TDM pad)
#define SB_LD (BN + 4)   // 132 halves/row (256B data + 8B TDM pad)

__global__ __launch_bounds__(256)
void CIA_gemm_bf16(const unsigned short* __restrict__ A, int lda,
                   const unsigned short* __restrict__ B, int ldb,
                   float* __restrict__ C, int ldc, int K) {
    __shared__ __align__(16) unsigned short sA[2][BM][SA_LD];
    __shared__ __align__(16) unsigned short sB[2][BK][SB_LD];

    const int tid    = threadIdx.x;
    const int lane   = tid & 31;
    const int ln     = lane & 15;
    const int hi     = lane >> 4;
    const int wave   = tid >> 5;
    const int waveM  = wave >> 1;          // 0..3
    const int waveN  = wave & 1;           // 0..1
    const int blockM = blockIdx.y * BM;
    const int blockN = blockIdx.x * BN;

    v8f acc[2][4] = {};
    const int nTiles = K / BK;

    if (wave == 0) {   // prologue: stage tile 0 into buffer 0
        tdm_load_tile_2d((unsigned)(size_t)&sA[0][0][0],
                         A + (size_t)blockM * lda,
                         (unsigned)lda, (unsigned long long)lda, BK, BM, 3, 3);
        tdm_load_tile_2d((unsigned)(size_t)&sB[0][0][0],
                         B + blockN,
                         (unsigned)ldb, (unsigned long long)ldb, BN, BK, 5, 1);
    }

    for (int t = 0; t < nTiles; ++t) {
        const int cur = t & 1;
        __syncthreads();   // everyone done reading the buffer we overwrite next
        if (wave == 0) {
            if (t + 1 < nTiles) {
                const int k0n = (t + 1) * BK;
                tdm_load_tile_2d((unsigned)(size_t)&sA[cur ^ 1][0][0],
                                 A + (size_t)blockM * lda + k0n,
                                 (unsigned)lda, (unsigned long long)lda, BK, BM, 3, 3);
                tdm_load_tile_2d((unsigned)(size_t)&sB[cur ^ 1][0][0],
                                 B + (size_t)k0n * ldb + blockN,
                                 (unsigned)ldb, (unsigned long long)ldb, BN, BK, 5, 1);
                __builtin_amdgcn_s_wait_tensorcnt(2);  // in-order => tile t done
            } else {
                __builtin_amdgcn_s_wait_tensorcnt(0);
            }
        }
        __syncthreads();   // tile t visible to all waves

        // ---- A fragments: K-contig rows, two ds_load_b128 per lane ----
        FragBF af[2], bfr[4];
#pragma unroll
        for (int mi = 0; mi < 2; ++mi) {
            const unsigned short* p = &sA[cur][waveM * 32 + mi * 16 + ln][0];
            af[mi].q[0] = *(const uint4*)(p + hi * 8);        // K = hi*8 .. +7
            af[mi].q[1] = *(const uint4*)(p + 16 + hi * 8);   // K = 16+hi*8 ..
        }
        // ---- B fragments: column n fixed, K down rows (same K order as A:
        //      halves[h] = K hi*8+h for h<8, 16+hi*8+(h-8) for h>=8) ----
#pragma unroll
        for (int ni = 0; ni < 4; ++ni) {
            const int n = waveN * 64 + ni * 16 + ln;
            const unsigned short* pb = &sB[cur][hi * 8][n];
#pragma unroll
            for (int h = 0; h < 16; ++h) bfr[ni].s[h] = pb[(h + (h & 8)) * SB_LD];
        }
#pragma unroll
        for (int mi = 0; mi < 2; ++mi)
#pragma unroll
            for (int ni = 0; ni < 4; ++ni)
                acc[mi][ni] = __builtin_amdgcn_wmma_f32_16x16x32_bf16(
                    false, af[mi].v, false, bfr[ni].v,
                    (short)0, acc[mi][ni], false, false);
    }

    // ---- epilogue: C/D layout -> lanes 0-15: M=r, lanes 16-31: M=8+r ----
#pragma unroll
    for (int mi = 0; mi < 2; ++mi)
#pragma unroll
        for (int ni = 0; ni < 4; ++ni) {
            int tr = blockM + waveM * 32 + mi * 16 + hi * 8;
            int tc = blockN + waveN * 64 + ni * 16 + ln;
            float* cp = C + (size_t)tr * ldc + tc;
#pragma unroll
            for (int r = 0; r < 8; ++r) cp[(size_t)r * ldc] = acc[mi][ni][r];
        }
}

// ---------------------------------------------------------------------------
// Row softmax: one workgroup (256 thr) per row; row staged in dynamic LDS
// (L<=16384 -> 64KB, well under the 320KB WGP LDS). Optional f32/bf16 out.
// ---------------------------------------------------------------------------
__global__ void CIA_softmax_rows(const float* __restrict__ in,
                                 float* __restrict__ outf,
                                 unsigned short* __restrict__ outb,
                                 int ld, int L) {
    extern __shared__ float buf[];
    __shared__ float red[8];
    const int tid  = threadIdx.x;
    const int lane = tid & 31;
    const int wv   = tid >> 5;
    const float* src = in + (size_t)blockIdx.x * ld;

    float mx = -3.402823466e38f;
    for (int i = tid * 4; i < L; i += 1024) {
        float4 v = *(const float4*)(src + i);
        buf[i] = v.x; buf[i + 1] = v.y; buf[i + 2] = v.z; buf[i + 3] = v.w;
        mx = fmaxf(mx, fmaxf(fmaxf(v.x, v.y), fmaxf(v.z, v.w)));
    }
    mx = waveRedMax(mx);
    if (lane == 0) red[wv] = mx;
    __syncthreads();
    float m = red[0];
#pragma unroll
    for (int i = 1; i < 8; ++i) m = fmaxf(m, red[i]);
    __syncthreads();

    float s = 0.f;
    for (int i = tid * 4; i < L; i += 1024) {
        float e0 = __expf(buf[i] - m),     e1 = __expf(buf[i + 1] - m);
        float e2 = __expf(buf[i + 2] - m), e3 = __expf(buf[i + 3] - m);
        buf[i] = e0; buf[i + 1] = e1; buf[i + 2] = e2; buf[i + 3] = e3;
        s += (e0 + e1) + (e2 + e3);
    }
    s = waveRedSum(s);
    if (lane == 0) red[wv] = s;
    __syncthreads();
    float tot = 0.f;
#pragma unroll
    for (int i = 0; i < 8; ++i) tot += red[i];
    const float inv = 1.0f / tot;

    float*          df = outf ? outf + (size_t)blockIdx.x * ld : nullptr;
    unsigned short* db = outb ? outb + (size_t)blockIdx.x * ld : nullptr;
    for (int i = tid * 4; i < L; i += 1024) {
        float o0 = buf[i] * inv, o1 = buf[i + 1] * inv;
        float o2 = buf[i + 2] * inv, o3 = buf[i + 3] * inv;
        if (df) { float4 v = make_float4(o0, o1, o2, o3); *(float4*)(df + i) = v; }
        if (db) {
            uint2 v;
            v.x = (unsigned)bf16bits(o0) | ((unsigned)bf16bits(o1) << 16);
            v.y = (unsigned)bf16bits(o2) | ((unsigned)bf16bits(o3) << 16);
            *(uint2*)(db + i) = v;
        }
    }
}

// ---------------------------------------------------------------------------
// Final: per (b,c,h) row of W=64: softmax over W + residual. One wave / row.
// ---------------------------------------------------------------------------
__global__ void CIA_final_softmax_w(const float* __restrict__ x,
                                    const float* __restrict__ y,
                                    float* __restrict__ out) {
    const int wave = (int)((blockIdx.x * blockDim.x + threadIdx.x) >> 5);
    const int lane = threadIdx.x & 31;
    const size_t base = (size_t)wave * 64;
    float a = y[base + lane], b = y[base + lane + 32];
    float mx = waveRedMax(fmaxf(a, b));
    float ea = __expf(a - mx), eb = __expf(b - mx);
    float s  = waveRedSum(ea + eb);
    float inv = 1.0f / s;
    out[base + lane]      = x[base + lane]      + ea * inv;
    out[base + lane + 32] = x[base + lane + 32] + eb * inv;
}

// ---------------------------------------------------------------------------
// Launch: ws layout = xbf(32MiB) | S(4MiB f32) | attn_bf(2MiB) | Y(64MiB f32)
// ---------------------------------------------------------------------------
extern "C" void kernel_launch(void* const* d_in, const int* in_sizes, int n_in,
                              void* d_out, int out_size, void* d_ws, size_t ws_size,
                              hipStream_t stream) {
    (void)in_sizes; (void)n_in; (void)out_size; (void)ws_size;
    const float* x = (const float*)d_in[0];
    float* out = (float*)d_out;

    char* w = (char*)d_ws;
    unsigned short* xbf   = (unsigned short*)w;                          // 33,554,432 B
    float*          S     = (float*)(w + 33554432u);                     //  4,194,304 B
    unsigned short* attnb = (unsigned short*)(w + 33554432u + 4194304u); //  2,097,152 B
    float*          Y     = (float*)(w + 33554432u + 4194304u + 2097152u); // 67,108,864 B

    // 1) x -> bf16 (flat buffer is simultaneously Xcn (1024x16384, ld 16384)
    //    and Xnc (16384x1024, ld 1024) — reference reshapes are raw views).
    CIA_cvt_bf16<<<TOTAL / (256 * 4), 256, 0, stream>>>(x, xbf);

    // 2) S = Xcn @ Xnc   (M=N=1024, K=16384)
    CIA_gemm_bf16<<<dim3(DIM_C / BN, DIM_C / BM), 256, 0, stream>>>(
        xbf, DIM_N, xbf, DIM_C, S, DIM_C, DIM_N);

    // 3) attn = softmax_rows(S) -> bf16
    CIA_softmax_rows<<<DIM_C, 256, DIM_C * sizeof(float), stream>>>(
        S, nullptr, attnb, DIM_C, DIM_C);

    // 4) Y = attn @ Xcn  (M=1024, N=16384, K=1024)
    CIA_gemm_bf16<<<dim3(DIM_N / BN, DIM_C / BM), 256, 0, stream>>>(
        attnb, DIM_C, xbf, DIM_N, Y, DIM_N, DIM_C);

    // 5) Y = softmax_rows(Y) in place (row = 16384 -> 64KB LDS stage)
    CIA_softmax_rows<<<DIM_C, 256, DIM_N * sizeof(float), stream>>>(
        Y, Y, nullptr, DIM_N, DIM_N);

    // 6) out = x + softmax_W(Y viewed (B,C,H,W))
    CIA_final_softmax_w<<<ROWS_W / 8, 256, 0, stream>>>(x, Y, out);
}
